// Model_60979945668671
// MI455X (gfx1250) — matrix operand order repaired and checked by reading, not compiled, not run on hardware
//
#include <hip/hip_runtime.h>
#include <hip/hip_bf16.h>

// Problem constants (from reference): B=64, S=200, NI=50000, H=256
#define B_   64
#define S_   200
#define T_   199
#define NI_  50000
#define H_   256
#define FH_  1024   // 4*H

typedef __attribute__((ext_vector_type(16))) __bf16 v16bf;
typedef __attribute__((ext_vector_type(8)))  float  v8f;

__device__ __forceinline__ unsigned short f2bf(float f) {
    union { float f; unsigned u; } v; v.f = f;
    unsigned u = v.u;
    u += 0x7FFFu + ((u >> 16) & 1u);   // round-to-nearest-even
    return (unsigned short)(u >> 16);
}

// ---------------------------------------------------------------------------
// Prep: convert Whh (f32 [1024,256]) into bf16 WMMA B-fragments.
// Fragment (w,g,c) is the 32x16 block  B[kk,nn] = Whh[256*g + 16*w + nn, 32*c + kk]
// in contiguous-K lane layout (same as h_frag): lanes 0-15 hold K=0..15 of column nn,
// lanes 16-31 hold K=16..31 -> each lane owns 16 consecutive K values (32 B).
// Also zeroes the global loss accumulators.
// ---------------------------------------------------------------------------
__global__ void prep_kernel(const float* __restrict__ Whh,
                            unsigned short* __restrict__ frags,
                            float* __restrict__ gaccum) {
    int fid  = blockIdx.x;        // 512 fragments: w[4b] g[2b] c[3b]
    int lane = threadIdx.x;       // 32 lanes
    int c = fid & 7;
    int g = (fid >> 3) & 3;
    int w = fid >> 5;
    int n = lane & 15;
    int khalf = (lane >> 4) << 4;          // 0 or 16
    int row = 256 * g + 16 * w + n;        // gate-output row of Whh
    int k0  = 32 * c + khalf;              // hidden-dim column of Whh
    #pragma unroll
    for (int p = 0; p < 16; ++p)
        frags[fid * 512 + lane * 16 + p] = f2bf(Whh[row * H_ + k0 + p]);
    if (fid == 0 && lane < 2) gaccum[lane] = 0.0f;
}

// ---------------------------------------------------------------------------
// Persistent fused LSTM + head + NLL. Grid: 4 blocks x 512 threads (16 waves).
// Block b owns batch rows [16b, 16b+16). Wave w owns hidden cols [16w, 16w+16)
// and computes the i/f/g/o gate tiles for that slice with bf16 WMMA.
// ---------------------------------------------------------------------------
__global__ __launch_bounds__(512) void lstm_kernel(
    const int*   __restrict__ items,
    const int*   __restrict__ actions,
    const float* __restrict__ WihT,      // [2*NI, 1024]
    const float* __restrict__ b_lstm,    // [1024]
    const float* __restrict__ trans_emb, // [NI, 256] (row 0 is zero)
    const float* __restrict__ Wq,        // [2, 256]
    const float* __restrict__ bq,        // [2]
    const unsigned short* __restrict__ frags,
    float* __restrict__ gaccum)          // [0]=loss sum, [1]=count
{
    // h in bf16, stored as WMMA A-fragments per 32-wide K chunk, contiguous K order:
    // lane L (<16) of chunk c holds row M=L, K = 0..15; lane L+16 holds
    // row M=L, K = 16..31. One contiguous 32B load per lane per chunk.
    __shared__ __align__(32) unsigned short h_frag[8 * 512];   // 8 KB
    __shared__ int   it_s[16], ac_s[16], q_s[16], tg_s[16];
    __shared__ float accum[16][2];

    const int tid  = threadIdx.x;
    const int lane = tid & 31;
    const int w    = tid >> 5;          // wave 0..15
    const int b0   = blockIdx.x * 16;
    const int l15  = lane & 15;
    const int hi   = lane >> 4;         // D-tile half: rows 0..7 vs 8..15
    const int jcol = 16 * w + l15;      // hidden column this lane holds in D tiles

    // per-lane constants
    float b4[4];
    #pragma unroll
    for (int g = 0; g < 4; ++g) b4[g] = b_lstm[256 * g + jcol];
    const float wq0 = Wq[jcol], wq1 = Wq[H_ + jcol];

    // where hidden column jcol lives inside the A-fragment layout
    const int chunk  = jcol >> 5;
    const int kp     = jcol & 31;
    const int lshift = ((kp >> 4) & 1) << 4;              // +16 lanes if K bit4
    const int slot   = kp & 15;                           // position within 16

    for (int i = tid; i < 8 * 512; i += 512) h_frag[i] = 0;   // h0 = 0

    float c_st[8];                                            // c in D-tile layout
    #pragma unroll
    for (int r = 0; r < 8; ++r) c_st[r] = 0.0f;

    float loss_loc = 0.0f, cnt_loc = 0.0f;

    for (int t = 0; t < T_; ++t) {
        if (tid < 16) {
            int b = b0 + tid;
            it_s[tid] = items  [b * S_ + t];
            ac_s[tid] = actions[b * S_ + t];
            q_s [tid] = items  [b * S_ + t + 1];
            tg_s[tid] = actions[b * S_ + t + 1];
        }
        if (tid < 32) ((float*)accum)[tid] = 0.0f;
        __syncthreads();

        // ---- load A (previous h, bf16) -----------------------------------
        v16bf a[8];
        #pragma unroll
        for (int c = 0; c < 8; ++c)
            a[c] = *reinterpret_cast<const v16bf*>(&h_frag[c * 512 + lane * 16]);
        __syncthreads();   // all h reads complete before this step's h writes

        // ---- C = x_proj + bias (gathered straight into D-tile layout) ----
        v8f acc[4];
        #pragma unroll
        for (int g = 0; g < 4; ++g) {
            #pragma unroll
            for (int r = 0; r < 8; ++r) {
                int m   = r + 8 * hi;
                int col = 256 * g + jcol;
                float xp = WihT[(long)it_s[m] * FH_ + col]
                         + WihT[(long)(NI_ + ac_s[m]) * FH_ + col];
                acc[g][r] = xp + b4[g];
            }
        }

        // ---- gates += h @ Whh^T : bf16 WMMA, fp32 accumulate -------------
        // Memory clobber per gate group: keeps the loop-invariant B-fragment
        // loads INSIDE the loop (Whh-bf16 is L2/WGP$-resident, 512 KB total)
        // instead of LICM-hoisting 256 VGPRs of weights and spilling to
        // scratch. Also caps in-flight B at 8 fragments (64 VGPRs).
        #pragma unroll
        for (int g = 0; g < 4; ++g) {
            asm volatile("" ::: "memory");
            v16bf bf[8];
            #pragma unroll
            for (int c = 0; c < 8; ++c) {
                int fid = (((w << 2) + g) << 3) | c;
                bf[c] = *reinterpret_cast<const v16bf*>(&frags[fid * 512 + lane * 16]);
            }
            #pragma unroll
            for (int c = 0; c < 8; ++c)
                acc[g] = __builtin_amdgcn_wmma_f32_16x16x32_bf16(
                             false, a[c], false, bf[c], (short)0, acc[g], false, false);
        }

        // ---- gate math, state update, fused head partials ----------------
        #pragma unroll
        for (int r = 0; r < 8; ++r) {
            int m = r + 8 * hi;
            float ig = 1.0f / (1.0f + __expf(-acc[0][r]));
            float fg = 1.0f / (1.0f + __expf(-acc[1][r]));
            float gg = tanhf(acc[2][r]);
            float og = 1.0f / (1.0f + __expf(-acc[3][r]));
            float cn = fg * c_st[r] + ig * gg;
            float hn = og * tanhf(cn);
            if (it_s[m] != 0) {                 // masked update
                c_st[r] = cn;
                h_frag[chunk * 512 + (m + lshift) * 16 + slot] = f2bf(hn);
                int q = q_s[m];                 // q!=0 implies mask true
                if (q != 0) {
                    float e = trans_emb[(long)q * H_ + jcol];
                    float p = hn * e;
                    atomicAdd(&accum[m][0], p * wq0);
                    atomicAdd(&accum[m][1], p * wq1);
                }
            }
        }
        __syncthreads();

        // ---- per-row 2-way log-softmax NLL -------------------------------
        if (tid < 16 && q_s[tid] != 0) {
            float q0 = accum[tid][0] + bq[0];
            float q1 = accum[tid][1] + bq[1];
            float mx = fmaxf(q0, q1);
            float lse = mx + __logf(__expf(q0 - mx) + __expf(q1 - mx));
            float chosen = (tg_s[tid] != 0) ? q1 : q0;
            loss_loc += lse - chosen;
            cnt_loc  += 1.0f;
        }
        __syncthreads();
    }

    if (tid < 16) {
        atomicAdd(&gaccum[0], loss_loc);
        atomicAdd(&gaccum[1], cnt_loc);
    }
}

__global__ void finalize_kernel(const float* __restrict__ gaccum, float* __restrict__ out) {
    if (threadIdx.x == 0)
        out[0] = (gaccum[1] > 0.0f) ? gaccum[0] / gaccum[1] : 0.0f;
}

// ---------------------------------------------------------------------------
extern "C" void kernel_launch(void* const* d_in, const int* in_sizes, int n_in,
                              void* d_out, int out_size, void* d_ws, size_t ws_size,
                              hipStream_t stream) {
    const int*   items     = (const int*)  d_in[0];
    const int*   actions   = (const int*)  d_in[1];
    const float* WihT      = (const float*)d_in[2];
    const float* Whh       = (const float*)d_in[3];
    const float* b_lstm    = (const float*)d_in[4];
    const float* trans_emb = (const float*)d_in[5];
    const float* Wq        = (const float*)d_in[6];
    const float* bq        = (const float*)d_in[7];

    float*          gaccum = (float*)d_ws;                               // 2 floats
    unsigned short* frags  = (unsigned short*)((char*)d_ws + 256);       // 512 KB bf16 Whh

    prep_kernel<<<512, 32, 0, stream>>>(Whh, frags, gaccum);
    lstm_kernel<<<4, 512, 0, stream>>>(items, actions, WihT, b_lstm,
                                       trans_emb, Wq, bq, frags, gaccum);
    finalize_kernel<<<1, 32, 0, stream>>>(gaccum, (float*)d_out);
}
